// DeformableConv_75256416960524
// MI455X (gfx1250) — matrix operand order repaired and checked by reading, not compiled, not run on hardware
//
#include <hip/hip_runtime.h>
#include <hip/hip_bf16.h>
#include <math.h>

// ---------------------------------------------------------------------------
// MI455X (gfx1250) implementation, round 2.
//
// Implicit-GEMM 3x3 conv on v_wmma_f32_16x16x32_f16.
//   M = out channels (64/block), N = pixels (64/block), K = 9*Cin (tap-major)
// 256 threads = 8 wave32s (4 M-tiles x 2 N-tiles); each wave owns TWO 16x16
// f32 accumulators (two N-subtiles) -> 2 wmma per K-chunk with A reuse.
//
// Round-2 codegen fixes:
//  * Weights pre-transposed once to f16 GEMM layout (padded to 64-row tiles):
//    A staging = 1 uint4 global load + 1 ds_store_b128 per thread, no branches.
//  * B tile stored in LDS in FRAGMENT-READY order: per-lane 16 halfs are
//    contiguous -> fragment read = ds_load_b128 x2 (was 16 ds_load_u16).
//  * Branchless clamped B-tile loads (no exec-mask serialization); each
//    thread covers 4 consecutive channels of one pixel, so the deformable
//    bilinear coords/weights are computed once per 4 gathers; B store is one
//    ds_store_b64.
//  * Kernel templated on <CIN, DEFORM>: chunk decomposition constant-folds.
//
// Workspace: ~89 MB (4 C-ch f32 tensors + offsets + gate + f16 weights).
// ---------------------------------------------------------------------------

typedef __attribute__((ext_vector_type(16))) _Float16 v16h;
typedef __attribute__((ext_vector_type(8)))  _Float16 v8h;
typedef __attribute__((ext_vector_type(4)))  _Float16 v4h;
typedef __attribute__((ext_vector_type(8)))  float    v8f;

#define HH 128
#define WW 128
#define BB 4
#define HWSZ (HH * WW)

enum { CF_RELU = 1, CF_RES = 2, CF_SIG = 4 };

__device__ __forceinline__ int iclamp(int v, int lo, int hi) {
    return v < lo ? lo : (v > hi ? hi : v);
}

template<int CIN, bool DEFORM>
__global__ __launch_bounds__(256, 2)
void conv3x3_wmma(const float* __restrict__ inA, const float* __restrict__ inB,
                  int CinA,
                  const _Float16* __restrict__ wt, int O,
                  const float* __restrict__ scale, const float* __restrict__ bias,
                  const float* __restrict__ resid,
                  float* __restrict__ outp, int Ostride,
                  const float* __restrict__ offs, int offBase,
                  int flags)
{
    constexpr int NCBLK  = CIN / 32;
    constexpr int NCHUNK = 9 * NCBLK;
    constexpr int K9     = 9 * CIN;

    __shared__ _Float16 As[64 * 32];   // weight slice  [64 oc][32 k]
    __shared__ _Float16 Bs[64 * 32];   // act tile, fragment-ready order

    const int tid  = threadIdx.x;
    const int lane = tid & 31;
    const int wid  = tid >> 5;
    const int wM   = wid & 3;            // 0..3
    const int wN   = wid >> 2;           // 0..1
    const int ln   = lane & 15;
    const int hs   = lane >> 4;

    // pixel tile: 64 contiguous x in one row
    constexpr int XT = WW / 64;
    const int bx = blockIdx.x;
    const int x0 = (bx % XT) * 64;
    const int y  = (bx / XT) % HH;
    const int b  = bx / (XT * HH);
    const int oT = blockIdx.y;

    v8f acc0 = {}, acc1 = {};

    // A-staging thread mapping (constant across chunks)
    const int ao = tid >> 2;             // 0..63 : out-channel row
    const int aq = tid & 3;              // 0..3  : 8-half quarter
    const size_t arow = (size_t)(oT * 64 + ao) * K9;

    for (int kc = 0; kc < NCHUNK; ++kc) {
        const int t    = kc / NCBLK;                 // constant-folded shifts
        const int cblk = kc - t * NCBLK;
        const int ky   = t / 3 - 1;
        const int kx   = t % 3 - 1;
        const int c0   = cblk * 32;

        __syncthreads();

        // ---- stage A: one 16B vector copy per thread (weights, f16) ----
        {
            const uint4* srcv = (const uint4*)(wt + arow + (size_t)t * CIN + c0 + aq * 8);
            *(uint4*)&As[ao * 32 + aq * 8] = *srcv;
        }

        // ---- stage B: 32 ch x 64 px, 4 channels per thread-iter ----
        #pragma unroll
        for (int i = 0; i < 2; ++i) {
            int idx = tid + i * 256;                 // 0..511
            int qk  = idx >> 6;                      // 0..7  (k quad)
            int p   = idx & 63;                      // pixel
            int k0  = qk * 4;
            int x   = x0 + p;
            v4h hv;
            if (!DEFORM) {
                const float* src; int cl0, CA;
                if (c0 < CinA) { src = inA; cl0 = c0;        CA = CinA; }
                else           { src = inB; cl0 = c0 - CinA; CA = CIN - CinA; }
                int  yy = y + ky, xx = x + kx;
                bool ok = ((unsigned)yy < (unsigned)HH) & ((unsigned)xx < (unsigned)WW);
                int  yyc = iclamp(yy, 0, HH - 1);
                int  xxc = iclamp(xx, 0, WW - 1);
                size_t rb = (((size_t)b * CA + cl0 + k0) * HH + yyc) * WW + xxc;
                #pragma unroll
                for (int cc = 0; cc < 4; ++cc) {
                    float v = src[rb + (size_t)cc * HWSZ];   // clamped addr, always safe
                    hv[cc] = (_Float16)(ok ? v : 0.0f);
                }
            } else {
                // bilinear coords shared by all 4 channels (group g = c0>>5)
                int g = c0 >> 5;
                size_t ob = (((size_t)b * 72 + offBase + (g * 9 + t) * 2) * HH + y) * WW + x;
                float dy = offs[ob];
                float dx = offs[ob + (size_t)HWSZ];
                float py = (float)(y + ky) + dy;
                float px = (float)(x + kx) + dx;
                float fy = floorf(py), fx = floorf(px);
                int   iy = (int)fy,    ix = (int)fx;
                float wy = py - fy,    wx = px - fx;
                float m00 = (((unsigned)iy     < (unsigned)HH) & ((unsigned)ix     < (unsigned)WW)) ? 1.0f : 0.0f;
                float m01 = (((unsigned)iy     < (unsigned)HH) & ((unsigned)(ix+1) < (unsigned)WW)) ? 1.0f : 0.0f;
                float m10 = (((unsigned)(iy+1) < (unsigned)HH) & ((unsigned)ix     < (unsigned)WW)) ? 1.0f : 0.0f;
                float m11 = (((unsigned)(iy+1) < (unsigned)HH) & ((unsigned)(ix+1) < (unsigned)WW)) ? 1.0f : 0.0f;
                int iy0 = iclamp(iy,     0, HH - 1), iy1 = iclamp(iy + 1, 0, HH - 1);
                int ix0 = iclamp(ix,     0, WW - 1), ix1 = iclamp(ix + 1, 0, WW - 1);
                float w00 = (1.0f - wy) * (1.0f - wx) * m00;
                float w01 = (1.0f - wy) * wx          * m01;
                float w10 = wy          * (1.0f - wx) * m10;
                float w11 = wy          * wx          * m11;
                int o00 = iy0 * WW + ix0, o01 = iy0 * WW + ix1;
                int o10 = iy1 * WW + ix0, o11 = iy1 * WW + ix1;
                size_t cb = ((size_t)b * CIN + c0 + k0) * HWSZ;
                #pragma unroll
                for (int cc = 0; cc < 4; ++cc) {
                    const float* base = inA + cb + (size_t)cc * HWSZ;
                    float v = base[o00] * w00 + base[o01] * w01
                            + base[o10] * w10 + base[o11] * w11;
                    hv[cc] = (_Float16)v;
                }
            }
            // fragment-ready slot: entry(k,p) = ((p>>4)*32 + (k>>4)*16 + (p&15))*16 + (k&15)
            int ent = (((p >> 4) * 32 + (k0 >> 4) * 16 + (p & 15)) * 16 + (k0 & 15));
            *(v4h*)&Bs[ent] = hv;                    // one ds_store_b64
        }

        // prefetch next weight slice (global_prefetch_b8)
        if (kc + 1 < NCHUNK) {
            int tn  = (kc + 1) / NCBLK;
            int cb2 = (kc + 1) - tn * NCBLK;
            __builtin_prefetch(wt + arow + (size_t)tn * CIN + cb2 * 32, 0, 0);
        }

        __syncthreads();

        // ---- fragments: all contiguous 16B LDS loads ----
        const int mrow = (wM * 16 + ln) * 32 + hs * 16;
        v8h alo = *(const v8h*)&As[mrow];            // K = hs*16 + 0..7
        v8h ahi = *(const v8h*)&As[mrow + 8];        // K = hs*16 + 8..15
        v16h a  = __builtin_shufflevector(alo, ahi, 0,1,2,3,4,5,6,7,
                                                     8,9,10,11,12,13,14,15);
        v16h b0 = *(const v16h*)&Bs[((wN * 2 + 0) * 32 + lane) * 16];
        v16h b1 = *(const v16h*)&Bs[((wN * 2 + 1) * 32 + lane) * 16];

        acc0 = __builtin_amdgcn_wmma_f32_16x16x32_f16(false, a, false, b0,
                                                      (short)0, acc0, false, false);
        acc1 = __builtin_amdgcn_wmma_f32_16x16x32_f16(false, a, false, b1,
                                                      (short)0, acc1, false, false);
    }

    // ---- fused epilogue (scale/bias, ReLU, residual, sigmoid) ----
    #pragma unroll
    for (int s = 0; s < 2; ++s) {
        v8f av = s ? acc1 : acc0;
        const int x = x0 + (wN * 2 + s) * 16 + ln;
        #pragma unroll
        for (int r = 0; r < 8; ++r) {
            int o = oT * 64 + wM * 16 + r + hs * 8;
            if (o < O) {
                float vv = av[r];
                float sc = scale ? scale[o] : 1.0f;
                float bb = bias  ? bias[o]  : 0.0f;
                vv = vv * sc + bb;
                if (flags & CF_RELU) vv = fmaxf(vv, 0.0f);
                size_t oi = (((size_t)b * Ostride + o) * HH + y) * WW + x;
                if (flags & CF_RES) vv += resid[oi];
                if (flags & CF_SIG) vv = 1.0f / (1.0f + __expf(-vv));
                outp[oi] = vv;
            }
        }
    }
}

// Transpose + convert weights to GEMM layout: Wt[o][t*Cin + ci] (f16),
// zero-padded to Opad rows so the conv A-stage needs no bounds checks.
__global__ void wtrans_kernel(const float* __restrict__ w, _Float16* __restrict__ wtout,
                              int O, int Opad, int Cin)
{
    int K9 = Cin * 9;
    int total = Opad * K9;
    for (int idx = blockIdx.x * 256 + threadIdx.x; idx < total; idx += gridDim.x * 256) {
        int o  = idx / K9;
        int k  = idx - o * K9;
        int t  = k / Cin;
        int ci = k - t * Cin;
        float v = (o < O) ? w[(size_t)o * K9 + (size_t)ci * 9 + t] : 0.0f;
        wtout[idx] = (_Float16)v;
    }
}

// Fold BN into per-channel scale/bias:  bn(conv+b) = conv*s + (b*s + be)
__global__ void make_sb_kernel(const float* __restrict__ g, const float* __restrict__ be,
                               const float* __restrict__ b,
                               float* __restrict__ s_out, float* __restrict__ b_out, int n)
{
    int i = blockIdx.x * blockDim.x + threadIdx.x;
    if (i < n) {
        float s = g[i] * 0.9999950000374997f;   // 1/sqrt(1 + 1e-5)
        s_out[i] = s;
        b_out[i] = b[i] * s + be[i];
    }
}

// out = vx[g] * out_x + vy[g] * out_y   (g = channel/32)
__global__ void combine_kernel(const float* __restrict__ v,
                               const float* __restrict__ ox,
                               const float* __restrict__ oy,
                               float* __restrict__ out)
{
    int idx = blockIdx.x * blockDim.x + threadIdx.x;
    const int N = BB * 64 * HWSZ;
    if (idx < N) {
        int hw = idx % HWSZ;
        int c  = (idx / HWSZ) & 63;
        int b  = idx / (64 * HWSZ);
        int g  = c >> 5;
        float vx = v[((size_t)b * 4 + g)     * HWSZ + hw];
        float vy = v[((size_t)b * 4 + 2 + g) * HWSZ + hw];
        out[idx] = vx * ox[idx] + vy * oy[idx];
    }
}

extern "C" void kernel_launch(void* const* d_in, const int* in_sizes, int n_in,
                              void* d_out, int out_size, void* d_ws, size_t ws_size,
                              hipStream_t stream)
{
    (void)in_sizes; (void)n_in; (void)out_size; (void)ws_size;

    const float* x = (const float*)d_in[0];
    const float* y = (const float*)d_in[1];
    const float* oc1_w   = (const float*)d_in[2];
    const float* oc1_b   = (const float*)d_in[3];
    const float* oc1_g   = (const float*)d_in[4];
    const float* oc1_be  = (const float*)d_in[5];
    const float* rbn_w1  = (const float*)d_in[6];
    const float* rbn_b1  = (const float*)d_in[7];
    const float* rbn_g1  = (const float*)d_in[8];
    const float* rbn_be1 = (const float*)d_in[9];
    const float* rbn_w2  = (const float*)d_in[10];
    const float* rbn_b2  = (const float*)d_in[11];
    const float* rbn_g2  = (const float*)d_in[12];
    const float* rbn_be2 = (const float*)d_in[13];
    const float* rb_w1   = (const float*)d_in[14];
    const float* rb_b1   = (const float*)d_in[15];
    const float* rb_w2   = (const float*)d_in[16];
    const float* rb_b2   = (const float*)d_in[17];
    const float* oc3_w   = (const float*)d_in[18];
    const float* oc3_b   = (const float*)d_in[19];
    const float* emb_w   = (const float*)d_in[20];
    const float* emb_b   = (const float*)d_in[21];
    const float* mm1_w   = (const float*)d_in[22];
    const float* mm1_b   = (const float*)d_in[23];
    const float* mmrb_w1 = (const float*)d_in[24];
    const float* mmrb_b1 = (const float*)d_in[25];
    const float* mmrb_w2 = (const float*)d_in[26];
    const float* mmrb_b2 = (const float*)d_in[27];
    const float* mm2_w   = (const float*)d_in[28];
    const float* mm2_b   = (const float*)d_in[29];
    const float* dcn_w   = (const float*)d_in[30];

    const size_t NT = (size_t)BB * 64 * HWSZ;          // 4,194,304 elems
    float* ws   = (float*)d_ws;
    float* sb   = ws;                                  // 1024 floats scale/bias
    float* t0   = ws + 1024;
    float* t1   = t0 + NT;
    float* t2   = t1 + NT;
    float* t3   = t2 + NT;
    float* offs = t3 + NT;                             // B*72*H*W
    float* vbuf = offs + (size_t)BB * 72 * HWSZ;       // B*4*H*W

    // f16 transposed-weight arena after the float region
    _Float16* wh = (_Float16*)(vbuf + (size_t)BB * 4 * HWSZ);
    const size_t W64  = 64 * 576;    // 64 rows x 9*64
    const size_t W128 = 64 * 1152;   // 64 rows x 9*128
    _Float16* wt_oc1   = wh;                       // 128-in
    _Float16* wt_rbn1  = wt_oc1   + W128;
    _Float16* wt_rbn2  = wt_rbn1  + W64;
    _Float16* wt_rb1   = wt_rbn2  + W64;
    _Float16* wt_rb2   = wt_rb1   + W64;
    _Float16* wt_oc3   = wt_rb2   + W64;           // padded to 128 rows
    _Float16* wt_dcn   = wt_oc3   + 2 * W64;
    _Float16* wt_emb   = wt_dcn   + W64;
    _Float16* wt_mm1   = wt_emb   + W64;           // 128-in
    _Float16* wt_mmrb1 = wt_mm1   + W128;
    _Float16* wt_mmrb2 = wt_mmrb1 + W64;
    _Float16* wt_mm2   = wt_mmrb2 + W64;           // padded to 64 rows

    float* out0 = (float*)d_out;
    float* outx = out0 + NT;
    float* outy = out0 + 2 * NT;

    dim3 blk(256);
    dim3 g1((WW / 64) * HH * BB, 1);               // 1024 pixel tiles
    dim3 g2((WW / 64) * HH * BB, 2);               // O=72 -> 2 M-tiles

    // ---- weight transforms (once per call, tiny) ----
    #define WT(dst, src, O_, Opad_, Cin_) \
        wtrans_kernel<<<((Opad_) * (Cin_) * 9 + 255) / 256, 256, 0, stream>>>(src, dst, O_, Opad_, Cin_)
    WT(wt_oc1,   oc1_w,   64, 64, 128);
    WT(wt_rbn1,  rbn_w1,  64, 64, 64);
    WT(wt_rbn2,  rbn_w2,  64, 64, 64);
    WT(wt_rb1,   rb_w1,   64, 64, 64);
    WT(wt_rb2,   rb_w2,   64, 64, 64);
    WT(wt_oc3,   oc3_w,   72, 128, 64);
    WT(wt_dcn,   dcn_w,   64, 64, 64);
    WT(wt_emb,   emb_w,   64, 64, 64);
    WT(wt_mm1,   mm1_w,   64, 64, 128);
    WT(wt_mmrb1, mmrb_w1, 64, 64, 64);
    WT(wt_mmrb2, mmrb_w2, 64, 64, 64);
    WT(wt_mm2,   mm2_w,    4, 64, 64);
    #undef WT

    // ---- BN fold ----
    make_sb_kernel<<<1, 64, 0, stream>>>(oc1_g,  oc1_be,  oc1_b,  sb + 0,   sb + 64,  64);
    make_sb_kernel<<<1, 64, 0, stream>>>(rbn_g1, rbn_be1, rbn_b1, sb + 128, sb + 192, 64);
    make_sb_kernel<<<1, 64, 0, stream>>>(rbn_g2, rbn_be2, rbn_b2, sb + 256, sb + 320, 64);

    // ---- offset branch ----
    conv3x3_wmma<128, false><<<g1, blk, 0, stream>>>(x, y, 64, wt_oc1, 64, sb + 0, sb + 64,
                                                     nullptr, t0, 64, nullptr, 0, 0);
    for (int i = 0; i < 5; ++i) {
        conv3x3_wmma<64, false><<<g1, blk, 0, stream>>>(t0, nullptr, 64, wt_rbn1, 64,
                                                        sb + 128, sb + 192, nullptr,
                                                        t2, 64, nullptr, 0, CF_RELU);
        conv3x3_wmma<64, false><<<g1, blk, 0, stream>>>(t2, nullptr, 64, wt_rbn2, 64,
                                                        sb + 256, sb + 320, t0,
                                                        t0, 64, nullptr, 0, CF_RES);
    }
    for (int i = 0; i < 3; ++i) {
        conv3x3_wmma<64, false><<<g1, blk, 0, stream>>>(t0, nullptr, 64, wt_rb1, 64,
                                                        nullptr, rb_b1, nullptr,
                                                        t2, 64, nullptr, 0, CF_RELU);
        conv3x3_wmma<64, false><<<g1, blk, 0, stream>>>(t2, nullptr, 64, wt_rb2, 64,
                                                        nullptr, rb_b2, t0,
                                                        t0, 64, nullptr, 0, CF_RES);
    }
    conv3x3_wmma<64, false><<<g2, blk, 0, stream>>>(t0, nullptr, 64, wt_oc3, 72,
                                                    nullptr, oc3_b, nullptr,
                                                    offs, 72, nullptr, 0, 0);

    // ---- deformable convs (bilinear gather fused into B-tile fill) ----
    conv3x3_wmma<64, true><<<g1, blk, 0, stream>>>(x, nullptr, 64, wt_dcn, 64,
                                                   nullptr, nullptr, nullptr,
                                                   outx, 64, offs, 0,  0);
    conv3x3_wmma<64, true><<<g1, blk, 0, stream>>>(y, nullptr, 64, wt_dcn, 64,
                                                   nullptr, nullptr, nullptr,
                                                   outy, 64, offs, 36, 0);

    // ---- embeddings + mask branch ----
    conv3x3_wmma<64, false><<<g1, blk, 0, stream>>>(outx, nullptr, 64, wt_emb, 64,
                                                    nullptr, emb_b, nullptr,
                                                    t2, 64, nullptr, 0, 0);
    conv3x3_wmma<64, false><<<g1, blk, 0, stream>>>(outy, nullptr, 64, wt_emb, 64,
                                                    nullptr, emb_b, nullptr,
                                                    t3, 64, nullptr, 0, 0);
    conv3x3_wmma<128, false><<<g1, blk, 0, stream>>>(t2, t3, 64, wt_mm1, 64,
                                                     nullptr, mm1_b, nullptr,
                                                     t0, 64, nullptr, 0, 0);
    for (int i = 0; i < 3; ++i) {
        conv3x3_wmma<64, false><<<g1, blk, 0, stream>>>(t0, nullptr, 64, wt_mmrb1, 64,
                                                        nullptr, mmrb_b1, nullptr,
                                                        t1, 64, nullptr, 0, CF_RELU);
        conv3x3_wmma<64, false><<<g1, blk, 0, stream>>>(t1, nullptr, 64, wt_mmrb2, 64,
                                                        nullptr, mmrb_b2, t0,
                                                        t0, 64, nullptr, 0, CF_RES);
    }
    conv3x3_wmma<64, false><<<g1, blk, 0, stream>>>(t0, nullptr, 64, wt_mm2, 4,
                                                    nullptr, mm2_b, nullptr,
                                                    vbuf, 4, nullptr, 0, CF_SIG);

    // ---- gated combine ----
    int N = (int)NT;
    combine_kernel<<<(N + 255) / 256, 256, 0, stream>>>(vbuf, outx, outy, out0);
}
